// DeepSimilarity_39548058862071
// MI455X (gfx1250) — matrix operand order repaired and checked by reading, not compiled, not run on hardware
//
#include <hip/hip_runtime.h>
#include <hip/hip_bf16.h>
#include <math.h>

typedef __attribute__((ext_vector_type(16))) _Float16 v16h;
typedef __attribute__((ext_vector_type(8)))  _Float16 v8h;
typedef __attribute__((ext_vector_type(8)))  float    v8f;

#define BQ   4
#define NQ   4096
#define MR   4096
#define NTOT (BQ*NQ)

#define TH1 (0.08f*0.08f)
#define TH2 (0.04f*0.04f)
#define TH3 (0.02f*0.02f)

// ---------------------------------------------------------------------------
// Kernel 1: ball query + gather of neighbor coordinates.
// One thread per query point. Reference points staged through LDS in tiles.
// ---------------------------------------------------------------------------
__global__ void ball_query_kernel(const float* __restrict__ pts,
                                  const float* __restrict__ knn,
                                  const float* __restrict__ img,
                                  float* __restrict__ realn1,
                                  float* __restrict__ realn2,
                                  float* __restrict__ realn3,
                                  float* __restrict__ imgn1)
{
    const int TS = 512;
    __shared__ float s[TS*4];
    const int tid = threadIdx.x;
    const int gq  = blockIdx.x*256 + tid;
    const int b   = gq / NQ;
    const int j   = gq % NQ;

    const float px = pts[(b*3+0)*NQ + j];
    const float py = pts[(b*3+1)*NQ + j];
    const float pz = pts[(b*3+2)*NQ + j];
    const float pp = px*px + py*py + pz*pz;

    const long base1 = (long)gq*64*3;
    const long base2 = (long)gq*32*3;
    const long base3 = (long)gq*16*3;

    {   // scan knnpoints, 3 thresholds
        const float* ref = knn + (long)b*3*MR;
        const float r0x = ref[0], r0y = ref[MR], r0z = ref[2*MR];
        int c1=0, c2=0, c3=0;
        float f1x=0,f1y=0,f1z=0, f2x=0,f2y=0,f2z=0, f3x=0,f3y=0,f3z=0;
        for (int t0=0; t0<MR; t0+=TS) {
            __syncthreads();
            for (int i=tid; i<TS; i+=256) {
                float x = ref[0*MR + t0+i];
                float y = ref[1*MR + t0+i];
                float z = ref[2*MR + t0+i];
                s[4*i+0]=x; s[4*i+1]=y; s[4*i+2]=z; s[4*i+3]=x*x+y*y+z*z;
            }
            __syncthreads();
            for (int t=0; t<TS; ++t) {
                float x=s[4*t+0], y=s[4*t+1], z=s[4*t+2], rr=s[4*t+3];
                float d2 = pp + rr - 2.0f*(px*x + py*y + pz*z);
                if (d2 < TH1 && c1 < 64) {
                    if (c1==0){f1x=x;f1y=y;f1z=z;}
                    realn1[base1+c1*3+0]=x; realn1[base1+c1*3+1]=y; realn1[base1+c1*3+2]=z; c1++;
                }
                if (d2 < TH2 && c2 < 32) {
                    if (c2==0){f2x=x;f2y=y;f2z=z;}
                    realn2[base2+c2*3+0]=x; realn2[base2+c2*3+1]=y; realn2[base2+c2*3+2]=z; c2++;
                }
                if (d2 < TH3 && c3 < 16) {
                    if (c3==0){f3x=x;f3y=y;f3z=z;}
                    realn3[base3+c3*3+0]=x; realn3[base3+c3*3+1]=y; realn3[base3+c3*3+2]=z; c3++;
                }
            }
        }
        if (c1==0){f1x=r0x;f1y=r0y;f1z=r0z;}
        if (c2==0){f2x=r0x;f2y=r0y;f2z=r0z;}
        if (c3==0){f3x=r0x;f3y=r0y;f3z=r0z;}
        for (int sl=c1; sl<64; ++sl){ realn1[base1+sl*3+0]=f1x; realn1[base1+sl*3+1]=f1y; realn1[base1+sl*3+2]=f1z; }
        for (int sl=c2; sl<32; ++sl){ realn2[base2+sl*3+0]=f2x; realn2[base2+sl*3+1]=f2y; realn2[base2+sl*3+2]=f2z; }
        for (int sl=c3; sl<16; ++sl){ realn3[base3+sl*3+0]=f3x; realn3[base3+sl*3+1]=f3y; realn3[base3+sl*3+2]=f3z; }
    }

    {   // scan imageneighbors, radius r, k=64
        const float* ref = img + (long)b*3*MR;
        const float r0x = ref[0], r0y = ref[MR], r0z = ref[2*MR];
        int c4=0; float f4x=0,f4y=0,f4z=0;
        for (int t0=0; t0<MR; t0+=TS) {
            __syncthreads();
            for (int i=tid; i<TS; i+=256) {
                float x = ref[0*MR + t0+i];
                float y = ref[1*MR + t0+i];
                float z = ref[2*MR + t0+i];
                s[4*i+0]=x; s[4*i+1]=y; s[4*i+2]=z; s[4*i+3]=x*x+y*y+z*z;
            }
            __syncthreads();
            for (int t=0; t<TS; ++t) {
                float x=s[4*t+0], y=s[4*t+1], z=s[4*t+2], rr=s[4*t+3];
                float d2 = pp + rr - 2.0f*(px*x + py*y + pz*z);
                if (d2 < TH1 && c4 < 64) {
                    if (c4==0){f4x=x;f4y=y;f4z=z;}
                    imgn1[base1+c4*3+0]=x; imgn1[base1+c4*3+1]=y; imgn1[base1+c4*3+2]=z; c4++;
                }
            }
        }
        if (c4==0){f4x=r0x;f4y=r0y;f4z=r0z;}
        for (int sl=c4; sl<64; ++sl){ imgn1[base1+sl*3+0]=f4x; imgn1[base1+sl*3+1]=f4y; imgn1[base1+sl*3+2]=f4z; }
    }
}

// ---------------------------------------------------------------------------
// Kernel 2: local similarity. 64 threads per (b,j) group, 4 groups per block.
// ---------------------------------------------------------------------------
__global__ void local_sim_kernel(const float* __restrict__ realn1,
                                 const float* __restrict__ imgn1,
                                 float* __restrict__ out)
{
    __shared__ float simg[4*64*4];
    __shared__ float ssum[256];
    const int tid = threadIdx.x;
    const int grp = tid >> 6;
    const int t   = tid & 63;
    const int g   = blockIdx.x*4 + grp;
    const long base = (long)g*64*3;

    float* si = simg + grp*256;
    {
        float x = imgn1[base+t*3+0], y = imgn1[base+t*3+1], z = imgn1[base+t*3+2];
        si[4*t+0]=x; si[4*t+1]=y; si[4*t+2]=z; si[4*t+3]=x*x+y*y+z*z;
    }
    float rx = realn1[base+t*3+0], ry = realn1[base+t*3+1], rz = realn1[base+t*3+2];
    float rr = rx*rx + ry*ry + rz*rz;
    __syncthreads();

    float mn = 3.4e38f;
    for (int l=0; l<64; ++l) {
        float d2 = rr + si[4*l+3] - 2.0f*(rx*si[4*l+0] + ry*si[4*l+1] + rz*si[4*l+2]);
        mn = fminf(mn, d2);
    }
    ssum[tid] = mn;
    __syncthreads();
    for (int st=32; st>0; st>>=1) {
        if (t < st) ssum[tid] += ssum[tid+st];
        __syncthreads();
    }
    if (t == 0) {
        float mean = ssum[tid] * (1.0f/64.0f);
        float sig  = 1.0f / (1.0f + expf(-mean));
        out[g] = 2.0f * (1.0f - sig);
    }
}

__global__ void fill_ones_kernel(float* __restrict__ out, int n)
{
    int i = blockIdx.x*256 + threadIdx.x;
    if (i < n) out[i] = 1.0f;
}

__global__ void zero_h_kernel(_Float16* __restrict__ p, long n)
{
    long i = (long)blockIdx.x*256 + threadIdx.x;
    if (i < n) p[i] = (_Float16)0.0f;
}

// ---------------------------------------------------------------------------
// Kernel 3: build repre in f16: repre[b][c*k+r][j] = A[c][r*(NQ/k)+j/k][j%k].
// chStride = channels per batch in output buffer (>= 7*k, pad stays zero).
// ---------------------------------------------------------------------------
__global__ void repre_kernel(const float* __restrict__ realn,
                             const float* __restrict__ pts,
                             _Float16* __restrict__ repre, int k, long chStride)
{
    const int gq = blockIdx.x*256 + threadIdx.x;
    const int b  = gq / NQ;
    const int j  = gq % NQ;
    const int q  = j / k;
    const int kk = j % k;
    const int stride = NQ / k;
    const float* P = pts + (long)b*3*NQ;
    _Float16* R = repre + (long)b*chStride*NQ;
    const long ck = (long)k*NQ;

    for (int r=0; r<k; ++r) {
        int nn = r*stride + q;
        long rb = ((long)(b*NQ + nn)*k + kk)*3;
        float x = realn[rb+0], y = realn[rb+1], z = realn[rb+2];
        float qx = P[nn], qy = P[NQ+nn], qz = P[2*NQ+nn];
        float dx = x-qx, dy = y-qy, dz = z-qz;
        float ds = dx*dx + dy*dy + dz*dz;
        long o = (long)r*NQ + j;
        R[o       ] = (_Float16)x;
        R[o + 1*ck] = (_Float16)y;
        R[o + 2*ck] = (_Float16)z;
        R[o + 3*ck] = (_Float16)dx;
        R[o + 4*ck] = (_Float16)dy;
        R[o + 5*ck] = (_Float16)dz;
        R[o + 6*ck] = (_Float16)ds;
    }
}

// f32 -> f16 weight conversion with zero-padding of K rows beyond C.
__global__ void f2h_pad_kernel(const float* __restrict__ s, _Float16* __restrict__ d,
                               int C, int D, int Kpad)
{
    int i = blockIdx.x*256 + threadIdx.x;
    if (i < Kpad*D) {
        int row = i / D;
        d[i] = (row < C) ? (_Float16)s[i] : (_Float16)0.0f;
    }
}

// ---------------------------------------------------------------------------
// Kernel 4: WMMA GEMM  Y(D x N) = act(W^T(D x K) * X(K x N) + bias)
// K is the padded depth (multiple of 32); X has K channels per batch, W has
// K rows (zero-padded). Tiles are staged into LDS in *fragment order* so each
// lane reads its whole 16-half WMMA operand as one aligned 32B LDS access.
// No guards -> no divergence anywhere (WMMA needs EXEC all-ones).
// ---------------------------------------------------------------------------
template<bool RELU, bool HALF_OUT>
__global__ void gemm_wmma_kernel(const _Float16* __restrict__ X,
                                 const _Float16* __restrict__ W,
                                 const float* __restrict__ bias,
                                 void* __restrict__ Yv,
                                 int K, int D,
                                 long xBatch, long yBatch)
{
    __shared__ __align__(32) _Float16 sA[2048];   // W tile, fragment order
    __shared__ __align__(32) _Float16 sB[2048];   // X tile, fragment order
    const int N    = NQ;
    const int tid  = threadIdx.x;
    const int lane = tid & 31;
    const int wave = tid >> 5;
    const int wpb  = blockDim.x >> 5;
    const int nthr = blockDim.x;
    const int m0b  = blockIdx.y * wpb * 16;
    const int n0   = blockIdx.x * 64;
    const int lm   = lane & 15;
    const int hi   = lane >> 4;
    const int dgs  = 2*wpb;                       // 8-col groups across block's D span
    const int dgsh = (wpb == 1) ? 1 : ((wpb == 2) ? 2 : 3);

    const _Float16* Xb = X + (long)blockIdx.z * xBatch;

    v8f acc[4] = {};
    for (int k0 = 0; k0 < K; k0 += 32) {
        __syncthreads();
        // ---- stage X tile (32 x 64) into sB[((t*2+hi)*16+lm)*16 + e] ----
        for (int u = tid; u < 256; u += nthr) {
            int kk = u >> 3, cg = u & 7;
            v8h v = *(const v8h*)(Xb + (long)(k0+kk)*N + n0 + cg*8);
            int thi = kk >> 4, e = kk & 15;
            #pragma unroll
            for (int jj=0; jj<8; ++jj) {
                int col = cg*8 + jj;
                int tt = col >> 4, lmm = col & 15;
                sB[(((tt*2 + thi)*16 + lmm) << 4) + e] = v[jj];
            }
        }
        // ---- stage W tile (32 x 16*wpb) into sA[((wv*2+hi)*16+lm)*16 + e] ----
        // A-layout (contiguous, matches B): kk -> hi=kk>>4, e=kk&15
        for (int u = tid; u < (dgs << 5); u += nthr) {
            int kk = u >> dgsh, dg = u & (dgs-1);
            v8h v = *(const v8h*)(W + (long)(k0+kk)*D + m0b + dg*8);
            int ahi = kk >> 4;
            int ae  = kk & 15;
            #pragma unroll
            for (int jj=0; jj<8; ++jj) {
                int dd = dg*8 + jj;
                int wv = dd >> 4, lmm = dd & 15;
                sA[(((wv*2 + ahi)*16 + lmm) << 4) + ae] = v[jj];
            }
        }
        __syncthreads();
        if (k0 + 32 < K)
            __builtin_prefetch(Xb + (long)(k0+32)*N + n0 + lane*4, 0, 1);

        v16h a = *(const v16h*)&sA[((wave*2 + hi)*16 + lm) << 4];
        #pragma unroll
        for (int t=0; t<4; ++t) {
            v16h bf = *(const v16h*)&sB[((t*2 + hi)*16 + lm) << 4];
            acc[t] = __builtin_amdgcn_wmma_f32_16x16x32_f16(
                false, a, false, bf, (short)0, acc[t], false, false);
        }
    }

    // Epilogue: C/D layout — lane: col = lm, VGPR r: row = m0 + r + 8*hi
    const int m0 = m0b + wave*16;
    #pragma unroll
    for (int t=0; t<4; ++t) {
        const int col = n0 + t*16 + lm;
        #pragma unroll
        for (int r=0; r<8; ++r) {
            int m = m0 + r + 8*hi;
            float v = acc[t][r] + bias[m];
            if (RELU) v = fmaxf(v, 0.0f);
            long o = (long)blockIdx.z*yBatch + (long)m*N + col;
            if (HALF_OUT) ((_Float16*)Yv)[o] = (_Float16)v;
            else          ((float*)Yv)[o]    = v;
        }
    }
}

// ---------------------------------------------------------------------------
extern "C" void kernel_launch(void* const* d_in, const int* in_sizes, int n_in,
                              void* d_out, int out_size, void* d_ws, size_t ws_size,
                              hipStream_t stream)
{
    const float* points = (const float*)d_in[0];
    const float* knn    = (const float*)d_in[1];
    const float* img    = (const float*)d_in[2];

    // Param leaves: handle both sorted-key order (conv first) and insertion
    // order (conv last); disambiguate via leaf sizes.
    const float* leaves[20];
    for (int i=0; i<20; ++i) leaves[i] = (const float*)d_in[4+i];
    const float *cW=nullptr, *cB=nullptr, *W1[3], *B1[3], *W2[3], *B2[3], *W3[3], *B3[3];
    int idx = 0;
    const bool convFirst = (in_sizes[4] == 160*64);
    if (convFirst) { cW = leaves[0]; cB = leaves[1]; idx = 2; }
    for (int i=0;i<3;++i){ W1[i]=leaves[idx++]; B1[i]=leaves[idx++]; }
    for (int i=0;i<3;++i){ W2[i]=leaves[idx++]; B2[i]=leaves[idx++]; }
    for (int i=0;i<3;++i){ W3[i]=leaves[idx++]; B3[i]=leaves[idx++]; }
    if (!convFirst) { cW = leaves[idx++]; cB = leaves[idx++]; }

    // Workspace layout (~90 MB; fits in 192 MB L2)
    char* wsb = (char*)d_ws;
    size_t off = 0;
    auto alloc = [&](size_t bytes) -> void* {
        void* r = wsb + off;
        off += (bytes + 255) & ~(size_t)255;
        return r;
    };
    float* realn1 = (float*)alloc((size_t)NTOT*64*3*sizeof(float));
    float* realn2 = (float*)alloc((size_t)NTOT*32*3*sizeof(float));
    float* realn3 = (float*)alloc((size_t)NTOT*16*3*sizeof(float));
    float* imgn1  = (float*)alloc((size_t)NTOT*64*3*sizeof(float));
    _Float16* repre1 = (_Float16*)alloc((size_t)BQ*448*NQ*2);
    _Float16* repre2 = (_Float16*)alloc((size_t)BQ*224*NQ*2);
    _Float16* repre3 = (_Float16*)alloc((size_t)BQ*128*NQ*2);   // 112 + 16 pad ch
    _Float16* bufA   = (_Float16*)alloc((size_t)BQ*448*NQ*2);
    _Float16* bufB   = (_Float16*)alloc((size_t)BQ*128*NQ*2);
    _Float16* bufP   = (_Float16*)alloc((size_t)BQ*128*NQ*2);   // mlp3 hidden, padded
    _Float16* feat   = (_Float16*)alloc((size_t)BQ*160*NQ*2);
    _Float16* hW1a = (_Float16*)alloc((size_t)448*448*2);
    _Float16* hW1b = (_Float16*)alloc((size_t)448*128*2);
    _Float16* hW1c = (_Float16*)alloc((size_t)128*64*2);
    _Float16* hW2a = (_Float16*)alloc((size_t)224*224*2);
    _Float16* hW2b = (_Float16*)alloc((size_t)224*128*2);
    _Float16* hW2c = (_Float16*)alloc((size_t)128*64*2);
    _Float16* hW3a = (_Float16*)alloc((size_t)128*112*2);       // Kpad=128
    _Float16* hW3b = (_Float16*)alloc((size_t)128*64*2);        // Kpad=128
    _Float16* hW3c = (_Float16*)alloc((size_t)64*32*2);
    _Float16* hWc  = (_Float16*)alloc((size_t)160*64*2);

    float* out_shallow = (float*)d_out;
    float* out_deep    = (float*)d_out + NTOT;
    float* out_feat    = (float*)d_out + 2*NTOT;

    // 1) geometry
    ball_query_kernel<<<NTOT/256, 256, 0, stream>>>(points, knn, img,
                                                    realn1, realn2, realn3, imgn1);
    // 2) similarity outputs
    local_sim_kernel<<<NTOT/4, 256, 0, stream>>>(realn1, imgn1, out_shallow);
    fill_ones_kernel<<<NTOT/256, 256, 0, stream>>>(out_deep, NTOT);
    // 3) zero padded buffers, then build repre (f16)
    {
        long nz = (long)BQ*128*NQ;
        zero_h_kernel<<<(int)((nz+255)/256), 256, 0, stream>>>(repre3, nz);
        zero_h_kernel<<<(int)((nz+255)/256), 256, 0, stream>>>(bufP,   nz);
    }
    repre_kernel<<<NTOT/256, 256, 0, stream>>>(realn1, points, repre1, 64, 448);
    repre_kernel<<<NTOT/256, 256, 0, stream>>>(realn2, points, repre2, 32, 224);
    repre_kernel<<<NTOT/256, 256, 0, stream>>>(realn3, points, repre3, 16, 128);
    // 4) weights -> f16 (with K zero-padding where needed)
    auto cvt = [&](const float* src, _Float16* dst, int C, int D, int Kpad) {
        int n = Kpad*D;
        f2h_pad_kernel<<<(n+255)/256, 256, 0, stream>>>(src, dst, C, D, Kpad);
    };
    cvt(W1[0], hW1a, 448, 448, 448); cvt(W1[1], hW1b, 448, 128, 448); cvt(W1[2], hW1c, 128, 64, 128);
    cvt(W2[0], hW2a, 224, 224, 224); cvt(W2[1], hW2b, 224, 128, 224); cvt(W2[2], hW2c, 128, 64, 128);
    cvt(W3[0], hW3a, 112, 112, 128); cvt(W3[1], hW3b, 112,  64, 128); cvt(W3[2], hW3c,  64, 32,  64);
    cvt(cW,    hWc,  160,  64, 160);
    // 5) WMMA GEMMs
    auto gemm = [&](const _Float16* X, const _Float16* Wp, const float* Bp, void* Y,
                    int K, int D, long xB, long yB, bool relu, bool halfOut) {
        int wpb = (D % 64 == 0) ? 4 : ((D % 32 == 0) ? 2 : 1);
        dim3 grid(NQ/64, D/(16*wpb), BQ);
        dim3 block(32*wpb);
        if (relu && halfOut)
            gemm_wmma_kernel<true, true ><<<grid, block, 0, stream>>>(X, Wp, Bp, Y, K, D, xB, yB);
        else if (relu && !halfOut)
            gemm_wmma_kernel<true, false><<<grid, block, 0, stream>>>(X, Wp, Bp, Y, K, D, xB, yB);
        else if (!relu && halfOut)
            gemm_wmma_kernel<false, true ><<<grid, block, 0, stream>>>(X, Wp, Bp, Y, K, D, xB, yB);
        else
            gemm_wmma_kernel<false, false><<<grid, block, 0, stream>>>(X, Wp, Bp, Y, K, D, xB, yB);
    };
    const long N = NQ;
    // mlp1: 448 -> 448 -> 128 -> 64   (feat channels 0..63)
    gemm(repre1, hW1a, B1[0], bufA,        448, 448, 448*N, 448*N, true,  true);
    gemm(bufA,   hW1b, B1[1], bufB,        448, 128, 448*N, 128*N, true,  true);
    gemm(bufB,   hW1c, B1[2], feat + 0*N,  128,  64, 128*N, 160*N, true,  true);
    // mlp2: 224 -> 224 -> 128 -> 64   (feat channels 64..127)
    gemm(repre2, hW2a, B2[0], bufA,        224, 224, 224*N, 224*N, true,  true);
    gemm(bufA,   hW2b, B2[1], bufB,        224, 128, 224*N, 128*N, true,  true);
    gemm(bufB,   hW2c, B2[2], feat + 64*N, 128,  64, 128*N, 160*N, true,  true);
    // mlp3 (K padded 112->128): 112 -> 112 -> 64 -> 32  (feat channels 128..159)
    gemm(repre3, hW3a, B3[0], bufP,        128, 112, 128*N, 128*N, true,  true);
    gemm(bufP,   hW3b, B3[1], bufB,        128,  64, 128*N,  64*N, true,  true);
    gemm(bufB,   hW3c, B3[2], feat +128*N,  64,  32,  64*N, 160*N, true,  true);
    // final conv: 160 -> 64, bias only, f32 straight into d_out
    gemm(feat,   hWc,  cB,   out_feat,     160,  64, 160*N,  64*N, false, false);

    (void)n_in; (void)out_size; (void)ws_size; (void)in_sizes;
}
